// DIEN_19172734009927
// MI455X (gfx1250) — matrix operand order repaired and checked by reading, not compiled, not run on hardware
//
#include <hip/hip_runtime.h>
#include <hip/hip_bf16.h>
#include <cstdint>
#include <cstddef>

typedef __attribute__((ext_vector_type(16))) _Float16 v16h;
typedef __attribute__((ext_vector_type(8)))  float    v8f;

#define BB   1024
#define TT   100
#define NEGK 5
#define U2d  128
#define N3d  384
#define AUXM (BB*(TT-1))          // 101376
#define BN_SCALE 0.99950037f      // 1/sqrt(1+1e-3)

__device__ __forceinline__ float sigf(float x) { return 1.f / (1.f + __expf(-x)); }

// ---------------- WMMA fragment helpers (16x16x32 f16 -> f32) ----------------
// A: 16x32 tile, row-major source [row][k]. lane%16 = M row; hi = lane/16.
// K slots: vgpr 0..3 -> k = 2i + 8*hi ; vgpr 4..7 -> k = 16 + 2(i-4) + 8*hi
__device__ __forceinline__ v16h load_fragA(const _Float16* base, int row, int ld,
                                           int k0, int hi) {
  union { v16h v; uint32_t u[8]; } f;
  const _Float16* p = base + (size_t)row * ld + k0 + 8 * hi;
#pragma unroll
  for (int i = 0; i < 4; ++i) f.u[i]     = *(const uint32_t*)(p + 2 * i);
#pragma unroll
  for (int i = 0; i < 4; ++i) f.u[4 + i] = *(const uint32_t*)(p + 16 + 2 * i);
  return f.v;
}
// B: 32x16 tile. Source is W^T row-major [n][k]. lane%16 = N col; hi = lane/16.
// K slots: vgpr 0..3 -> k = 2i + 8*hi ; vgpr 4..7 -> k = 16 + 2(i-4) + 8*hi
__device__ __forceinline__ v16h load_fragB(const _Float16* base, int col, int ld,
                                           int k0, int hi) {
  union { v16h v; uint32_t u[8]; } f;
  const _Float16* p = base + (size_t)col * ld + k0 + 8 * hi;
#pragma unroll
  for (int i = 0; i < 4; ++i) f.u[i]     = *(const uint32_t*)(p + 2 * i);
#pragma unroll
  for (int i = 0; i < 4; ++i) f.u[4 + i] = *(const uint32_t*)(p + 16 + 2 * i);
  return f.v;
}
__device__ __forceinline__ v8f wmma_f16(v16h a, v16h b, v8f c) {
  return __builtin_amdgcn_wmma_f32_16x16x32_f16(false, a, false, b, (short)0, c,
                                                false, false);
}

// ---------------- generic utility kernels ----------------
__global__ void zero_f32(float* p, int n) {
  int i = blockIdx.x * blockDim.x + threadIdx.x;
  if (i < n) p[i] = 0.f;
}
__global__ void zero_f16(_Float16* p, int n) {
  int i = blockIdx.x * blockDim.x + threadIdx.x;
  if (i < n) p[i] = (_Float16)0.f;
}
// W [K][N] f32 -> Wt [Npad][Kpad] f16 (zero padded)
__global__ void convT(const float* __restrict__ W, _Float16* __restrict__ Wt,
                      int K, int N, int Kpad, int Npad) {
  int i = blockIdx.x * blockDim.x + threadIdx.x;
  if (i >= Npad * Kpad) return;
  int n = i / Kpad, k = i % Kpad;
  float v = (n < N && k < K) ? W[(size_t)k * N + n] : 0.f;
  Wt[i] = (_Float16)v;
}

// ---------------- embedding gather + sums ----------------
__global__ __launch_bounds__(128) void gather_kernel(
    const int* __restrict__ user, const int* __restrict__ item,
    const int* __restrict__ cate, const int* __restrict__ item_his,
    const int* __restrict__ cate_his, const int* __restrict__ no_m,
    const int* __restrict__ no_c, const float* __restrict__ ut,
    const float* __restrict__ it, const float* __restrict__ ct,
    float* __restrict__ user_emb, float* __restrict__ item_emb,
    float* __restrict__ his_emb, _Float16* __restrict__ his_emb_h,
    float* __restrict__ his_sum, float* __restrict__ noclk_sum,
    float* __restrict__ noclk_his) {
  int b = blockIdx.x, j = threadIdx.x;  // 128 threads
  if (j < 64) user_emb[b * 64 + j] = ut[(size_t)user[b] * 64 + j];
  item_emb[b * 128 + j] = (j < 64) ? it[(size_t)item[b] * 64 + j]
                                   : ct[(size_t)cate[b] * 64 + (j - 64)];
  float hsum = 0.f;
  for (int t = 0; t < TT; ++t) {
    int idx = (j < 64) ? item_his[b * TT + t] : cate_his[b * TT + t];
    float v = it[(size_t)idx * 64 + (j & 63)];
    size_t o = ((size_t)b * TT + t) * 128 + j;
    his_emb[o] = v;
    his_emb_h[o] = (_Float16)v;
    hsum += v;
  }
  his_sum[b * 128 + j] = hsum;
  float nsum = 0.f;
  for (int t = 0; t < TT; ++t) {
    for (int k = 0; k < NEGK; ++k) {
      int idx = (j < 64) ? no_m[(b * TT + t) * NEGK + k]
                         : no_c[(b * TT + t) * NEGK + k];
      float v = it[(size_t)idx * 64 + (j & 63)];
      nsum += v;
      if (k == 0) noclk_his[((size_t)b * TT + t) * 128 + j] = v;
    }
  }
  noclk_sum[b * 128 + j] = nsum;
}

// ---------------- tiled WMMA GEMM: C = act(A(f16 MxK) * Wt^T + bias) ----------
__global__ __launch_bounds__(128) void gemm_f16(
    const _Float16* __restrict__ A, const _Float16* __restrict__ Wt,
    const float* __restrict__ bias, float* __restrict__ Cf,
    _Float16* __restrict__ Ch, int M, int K, int Npad, int Nreal, int act) {
  int lane = threadIdx.x, wave = threadIdx.y;  // (32,4)
  int m0 = blockIdx.x * 16;
  int n0 = (blockIdx.y * 4 + wave) * 16;
  if (n0 >= Npad) return;
  int l16 = lane & 15, hi = lane >> 4;
  v8f c;
#pragma unroll
  for (int i = 0; i < 8; ++i) c[i] = 0.f;
  for (int k0 = 0; k0 < K; k0 += 32) {
    v16h a = load_fragA(A, m0 + l16, K, k0, hi);
    v16h b = load_fragB(Wt, n0 + l16, K, k0, hi);
    c = wmma_f16(a, b, c);
  }
  int n = n0 + l16;
  float bv = (bias && n < Nreal) ? bias[n] : 0.f;
#pragma unroll
  for (int r = 0; r < 8; ++r) {
    int m = m0 + r + 8 * hi;
    float v = c[r] + bv;
    if (act == 1) v = sigf(v);
    size_t o = (size_t)m * Npad + n;
    if (Ch) Ch[o] = (_Float16)v;
    else    Cf[o] = v;
  }
}

// ---------------- GRU recurrence (16 batch rows / block, WMMA per step) ------
__global__ __launch_bounds__(256) void gru_kernel(
    const float* __restrict__ xp,     // [B*T*384], includes b[0]
    const _Float16* __restrict__ whT, // [384][128]
    const float* __restrict__ bh,     // b[1], 384
    const int* __restrict__ mask, float* __restrict__ gru_out,
    _Float16* __restrict__ gru_out_h) {
  __shared__ float    hs[16 * 128];
  __shared__ _Float16 h16[16 * 128];
  __shared__ float    rec[16 * 384];
  int lane = threadIdx.x, wave = threadIdx.y;  // (32,8)
  int tid = wave * 32 + lane;
  int b0 = blockIdx.x * 16;
  int l16 = lane & 15, hi = lane >> 4;
  for (int i = tid; i < 16 * 128; i += 256) hs[i] = 0.f;
  __syncthreads();
  for (int t = 0; t < TT; ++t) {
    for (int i = tid; i < 16 * 128; i += 256) h16[i] = (_Float16)hs[i];
    __syncthreads();
    // rec = h @ wh + bh : 24 N-tiles over 8 waves
#pragma unroll
    for (int q = 0; q < 3; ++q) {
      int n0 = (wave * 3 + q) * 16;
      v8f c;
#pragma unroll
      for (int i = 0; i < 8; ++i) c[i] = 0.f;
#pragma unroll
      for (int kt = 0; kt < 4; ++kt) {
        v16h a = load_fragA(h16, l16, 128, kt * 32, hi);
        v16h b = load_fragB(whT, n0 + l16, 128, kt * 32, hi);
        c = wmma_f16(a, b, c);
      }
      int n = n0 + l16;
#pragma unroll
      for (int r = 0; r < 8; ++r)
        rec[(r + 8 * hi) * 384 + n] = c[r] + bh[n];
    }
    __syncthreads();
    for (int e = tid; e < 16 * 128; e += 256) {
      int m = e >> 7, j = e & 127;
      int brow = b0 + m;
      size_t base = ((size_t)brow * TT + t) * 384;
      float xz = xp[base + j], xr = xp[base + 128 + j], xh = xp[base + 256 + j];
      float rz = rec[m * 384 + j], rr = rec[m * 384 + 128 + j],
            rh = rec[m * 384 + 256 + j];
      float h_old = hs[e];
      float z = sigf(xz + rz), r = sigf(xr + rr);
      float hh = tanhf(xh + r * rh);
      float hn = z * h_old + (1.f - z) * hh;
      if (mask[brow * TT + t] <= 0) hn = h_old;
      hs[e] = hn;
      size_t ob = ((size_t)brow * TT + t) * 128 + j;
      gru_out[ob] = hn;
      gru_out_h[ob] = (_Float16)hn;
    }
    __syncthreads();
  }
}

// ---------------- AUGRU (att==1 after size-1 softmax); hh uses (r*s)@ws_h ----
__global__ __launch_bounds__(256) void augru_kernel(
    const float* __restrict__ xp,     // [B*T*384], includes b
    const _Float16* __restrict__ wsT, // [384][128]; rows 0..255 = r,u; 256..383 = h
    const int* __restrict__ mask, float* __restrict__ behavior) {
  __shared__ float    ss[16 * 128];
  __shared__ _Float16 s16[16 * 128];
  __shared__ _Float16 rs16[16 * 128];
  __shared__ float    us[16 * 128];
  __shared__ float    rec[16 * 256];
  __shared__ float    rech[16 * 128];
  int lane = threadIdx.x, wave = threadIdx.y;  // (32,8)
  int tid = wave * 32 + lane;
  int b0 = blockIdx.x * 16;
  int l16 = lane & 15, hi = lane >> 4;
  for (int i = tid; i < 16 * 128; i += 256) ss[i] = 0.f;
  __syncthreads();
  for (int t = 0; t < TT; ++t) {
    for (int i = tid; i < 16 * 128; i += 256) s16[i] = (_Float16)ss[i];
    __syncthreads();
    // phase 1: rec = s @ ws[:, 0:256]  (16 tiles over 8 waves)
#pragma unroll
    for (int q = 0; q < 2; ++q) {
      int n0 = (wave * 2 + q) * 16;
      v8f c;
#pragma unroll
      for (int i = 0; i < 8; ++i) c[i] = 0.f;
#pragma unroll
      for (int kt = 0; kt < 4; ++kt) {
        v16h a = load_fragA(s16, l16, 128, kt * 32, hi);
        v16h b = load_fragB(wsT, n0 + l16, 128, kt * 32, hi);
        c = wmma_f16(a, b, c);
      }
      int n = n0 + l16;
#pragma unroll
      for (int r = 0; r < 8; ++r) rec[(r + 8 * hi) * 256 + n] = c[r];
    }
    __syncthreads();
    for (int e = tid; e < 16 * 128; e += 256) {
      int m = e >> 7, j = e & 127;
      size_t base = ((size_t)(b0 + m) * TT + t) * 384;
      float r = sigf(xp[base + j] + rec[m * 256 + j]);
      float u = sigf(xp[base + 128 + j] + rec[m * 256 + 128 + j]);
      us[e] = u;
      rs16[e] = (_Float16)(r * ss[e]);
    }
    __syncthreads();
    // phase 2: rech = (r*s) @ ws[:, 256:384]  (8 tiles over 8 waves)
    {
      int n0 = wave * 16;
      v8f c;
#pragma unroll
      for (int i = 0; i < 8; ++i) c[i] = 0.f;
#pragma unroll
      for (int kt = 0; kt < 4; ++kt) {
        v16h a = load_fragA(rs16, l16, 128, kt * 32, hi);
        v16h b = load_fragB(wsT, 256 + n0 + l16, 128, kt * 32, hi);
        c = wmma_f16(a, b, c);
      }
      int n = n0 + l16;
#pragma unroll
      for (int r = 0; r < 8; ++r) rech[(r + 8 * hi) * 128 + n] = c[r];
    }
    __syncthreads();
    for (int e = tid; e < 16 * 128; e += 256) {
      int m = e >> 7, j = e & 127;
      int brow = b0 + m;
      size_t base = ((size_t)brow * TT + t) * 384;
      float hh = tanhf(xp[base + 256 + j] + rech[e]);
      float u = us[e];  // au = att*u with att == 1
      float hn = (1.f - u) * ss[e] + u * hh;
      if (mask[brow * TT + t] <= 0) hn = ss[e];
      ss[e] = hn;
    }
    __syncthreads();
  }
  for (int e = tid; e < 16 * 128; e += 256) {
    int m = e >> 7, j = e & 127;
    behavior[(size_t)(b0 + m) * 128 + j] = ss[e];
  }
}

// ---------------- aux MLP input build (applies BN, outputs f16) --------------
__global__ __launch_bounds__(256) void aux_build(
    const float* __restrict__ gru_out, const float* __restrict__ his_emb,
    const float* __restrict__ noclk_his, const float* __restrict__ gamma,
    const float* __restrict__ beta, _Float16* __restrict__ clk_in,
    _Float16* __restrict__ nclk_in) {
  int row = blockIdx.x;        // 0 .. 101375
  int f = threadIdx.x;         // 0 .. 255
  int b = row / (TT - 1), t = row % (TT - 1);
  float g = gamma[f] * BN_SCALE, be = beta[f];
  float sc, sn;
  if (f < 128) {
    sc = gru_out[((size_t)b * TT + t) * 128 + f];
    sn = sc;
  } else {
    sc = his_emb[((size_t)b * TT + t + 1) * 128 + (f - 128)];
    sn = noclk_his[((size_t)b * TT + t + 1) * 128 + (f - 128)];
  }
  size_t o = (size_t)row * 256 + f;
  clk_in[o] = (_Float16)(g * sc + be);
  nclk_in[o] = (_Float16)(g * sn + be);
}

// ---------------- aux layers 2+3 + loss reduction ----------------------------
__global__ __launch_bounds__(256) void aux_tail(
    const _Float16* __restrict__ clk_h1, const _Float16* __restrict__ nclk_h1,
    const float* __restrict__ w2, const float* __restrict__ b2,
    const float* __restrict__ w3, const float* __restrict__ b3,
    const int* __restrict__ mask, float* __restrict__ acc) {
  __shared__ float w2s[100 * 50];
  __shared__ float w3s[50];
  __shared__ float red[256];
  int tid = threadIdx.x;
  for (int i = tid; i < 100 * 50; i += 256) w2s[i] = w2[i];
  if (tid < 50) w3s[tid] = w3[tid];
  __syncthreads();
  int idx = blockIdx.x * 256 + tid;
  float loss = 0.f;
  if (idx < 2 * AUXM) {
    int which = idx >= AUXM;
    int row = idx - which * AUXM;
    const _Float16* h1 = (which ? nclk_h1 : clk_h1) + (size_t)row * 112;
    float h2[50];
#pragma unroll 1
    for (int j = 0; j < 50; ++j) {
      float a = b2[j];
      for (int i = 0; i < 100; ++i) a += (float)h1[i] * w2s[i * 50 + j];
      h2[j] = sigf(a);
    }
    float o = b3[0];
#pragma unroll
    for (int j = 0; j < 50; ++j) o += h2[j] * w3s[j];
    float p = sigf(o);
    int b = row / (TT - 1), t = row % (TT - 1);
    float mf = (mask[b * TT + t + 1] > 0) ? 1.f : 0.f;
    loss = which ? (-logf(1.f - p) * mf) : (-logf(p) * mf);
  }
  red[tid] = loss;
  __syncthreads();
  for (int st = 128; st > 0; st >>= 1) {
    if (tid < st) red[tid] += red[tid + st];
    __syncthreads();
  }
  if (tid == 0) atomicAdd(acc, red[0]);
}

// ---------------- final-FC input build (BN applied, f16) ---------------------
__global__ __launch_bounds__(256) void fc_build(
    const float* __restrict__ user_emb, const float* __restrict__ item_emb,
    const float* __restrict__ his_sum, const float* __restrict__ behavior,
    const float* __restrict__ noclk_sum, const float* __restrict__ gamma,
    const float* __restrict__ beta, _Float16* __restrict__ fcin) {
  int idx = blockIdx.x * 256 + threadIdx.x;  // 1024*576
  if (idx >= BB * 576) return;
  int b = idx / 576, f = idx % 576;
  float s;
  if (f < 64)       s = user_emb[b * 64 + f];
  else if (f < 192) s = item_emb[b * 128 + (f - 64)];
  else if (f < 320) s = his_sum[b * 128 + (f - 192)];
  else if (f < 448) s = behavior[b * 128 + (f - 320)];
  else              s = noclk_sum[b * 128 + (f - 448)];
  fcin[idx] = (_Float16)(gamma[f] * BN_SCALE * s + beta[f]);
}

// ---------------- DICE: batch mean/var per column, then gated output ---------
__global__ __launch_bounds__(256) void dice_kernel(
    const float* __restrict__ A, int lda, const float* __restrict__ alpha,
    _Float16* __restrict__ outh, int ldo) {
  __shared__ float s1[256], s2[256];
  int c = blockIdx.x, tid = threadIdx.x;
  float sm = 0.f, sq = 0.f;
  for (int r = tid; r < BB; r += 256) {
    float x = A[(size_t)r * lda + c];
    sm += x;
    sq += x * x;
  }
  s1[tid] = sm;
  s2[tid] = sq;
  __syncthreads();
  for (int st = 128; st > 0; st >>= 1) {
    if (tid < st) { s1[tid] += s1[tid + st]; s2[tid] += s2[tid + st]; }
    __syncthreads();
  }
  float mean = s1[0] / (float)BB;
  float var = s2[0] / (float)BB - mean * mean;
  float rs = rsqrtf(var + 0.001f);
  float al = alpha[c];
  for (int r = tid; r < BB; r += 256) {
    float x = A[(size_t)r * lda + c];
    float p = sigf((x - mean) * rs);
    outh[(size_t)r * ldo + c] = (_Float16)(p * x + al * (1.f - p) * x);
  }
}

// ---------------- final layer3 + softmax + aux_loss write --------------------
__global__ __launch_bounds__(256) void final_kernel(
    const _Float16* __restrict__ h2, const float* __restrict__ w3,
    const float* __restrict__ b3, const float* __restrict__ acc,
    float* __restrict__ out) {
  int b = blockIdx.x * 256 + threadIdx.x;
  if (b < BB) {
    const _Float16* h = h2 + (size_t)b * 96;
    float o0 = b3[0], o1 = b3[1];
#pragma unroll 1
    for (int k = 0; k < 80; ++k) {
      float v = (float)h[k];
      o0 += v * w3[k * 2];
      o1 += v * w3[k * 2 + 1];
    }
    out[b * 2] = o0;
    out[b * 2 + 1] = o1;
    float m = fmaxf(o0, o1);
    float e0 = expf(o0 - m), e1 = expf(o1 - m);
    float s = e0 + e1;
    out[2 * BB + b * 2] = e0 / s;
    out[2 * BB + b * 2 + 1] = e1 / s;
  }
  if (b == 0) out[4 * BB] = acc[0] / (float)(2 * 0 + AUXM);  // mean over (B,99)
}

// =============================================================================
extern "C" void kernel_launch(void* const* d_in, const int* in_sizes, int n_in,
                              void* d_out, int out_size, void* d_ws,
                              size_t ws_size, hipStream_t stream) {
  (void)in_sizes; (void)n_in; (void)out_size; (void)ws_size;
  // ---- inputs (setup_inputs dict order, params flattened in insertion order)
  const int* user     = (const int*)d_in[0];
  const int* item     = (const int*)d_in[1];
  const int* cate     = (const int*)d_in[2];
  const int* item_his = (const int*)d_in[3];
  const int* cate_his = (const int*)d_in[4];
  const int* mask     = (const int*)d_in[5];
  const int* no_m     = (const int*)d_in[6];
  const int* no_c     = (const int*)d_in[7];
  const float* ut       = (const float*)d_in[8];
  const float* it       = (const float*)d_in[9];
  const float* ct       = (const float*)d_in[10];
  const float* gru_wx   = (const float*)d_in[11];
  const float* gru_wh   = (const float*)d_in[12];
  const float* gru_b    = (const float*)d_in[13];   // (2,384)
  const float* aug_wx   = (const float*)d_in[14];
  const float* aug_ws   = (const float*)d_in[15];
  const float* aug_b    = (const float*)d_in[16];   // (384,)
  const float* aux_g    = (const float*)d_in[17];
  const float* aux_be   = (const float*)d_in[18];
  const float* aux_w1   = (const float*)d_in[19];
  const float* aux_b1   = (const float*)d_in[20];
  const float* aux_w2   = (const float*)d_in[21];
  const float* aux_b2   = (const float*)d_in[22];
  const float* aux_w3   = (const float*)d_in[23];
  const float* aux_b3   = (const float*)d_in[24];
  // d_in[25..32]: att params — dead code (softmax over size-1 axis == 1.0)
  const float* fc_g     = (const float*)d_in[33];
  const float* fc_be    = (const float*)d_in[34];
  const float* fc_w1    = (const float*)d_in[35];
  const float* fc_b1    = (const float*)d_in[36];
  const float* fc_a1    = (const float*)d_in[37];
  const float* fc_w2    = (const float*)d_in[38];
  const float* fc_b2    = (const float*)d_in[39];
  const float* fc_a2    = (const float*)d_in[40];
  const float* fc_w3    = (const float*)d_in[41];
  const float* fc_b3    = (const float*)d_in[42];
  float* out = (float*)d_out;

  // ---- workspace carve-out
  size_t off = 0;
  char* wsb = (char*)d_ws;
  auto alloc = [&](size_t bytes) -> void* {
    void* p = wsb + off;
    off += (bytes + 255) & ~(size_t)255;
    return p;
  };
  const size_t BT128 = (size_t)BB * TT * 128, BT384 = (size_t)BB * TT * 384;
  float* his_emb   = (float*)alloc(BT128 * 4);
  float* noclk_his = (float*)alloc(BT128 * 4);
  float* gru_xp    = (float*)alloc(BT384 * 4);
  float* aug_xp    = (float*)alloc(BT384 * 4);
  float* gru_out   = (float*)alloc(BT128 * 4);
  float* his_sum   = (float*)alloc((size_t)BB * 128 * 4);
  float* noclk_sum = (float*)alloc((size_t)BB * 128 * 4);
  float* user_emb  = (float*)alloc((size_t)BB * 64 * 4);
  float* item_emb  = (float*)alloc((size_t)BB * 128 * 4);
  float* behavior  = (float*)alloc((size_t)BB * 128 * 4);
  float* a1        = (float*)alloc((size_t)BB * 208 * 4);
  float* a2        = (float*)alloc((size_t)BB * 80 * 4);
  float* acc       = (float*)alloc(256);
  _Float16* his_emb_h = (_Float16*)alloc(BT128 * 2);
  _Float16* gru_out_h = (_Float16*)alloc(BT128 * 2);
  _Float16* clk_in  = (_Float16*)alloc((size_t)AUXM * 256 * 2);
  _Float16* nclk_in = (_Float16*)alloc((size_t)AUXM * 256 * 2);
  _Float16* clk_h1  = (_Float16*)alloc((size_t)AUXM * 112 * 2);
  _Float16* nclk_h1 = (_Float16*)alloc((size_t)AUXM * 112 * 2);
  _Float16* fcin_h  = (_Float16*)alloc((size_t)BB * 576 * 2);
  _Float16* h1_h    = (_Float16*)alloc((size_t)BB * 224 * 2);
  _Float16* h2_h    = (_Float16*)alloc((size_t)BB * 96 * 2);
  _Float16* wxT_gru = (_Float16*)alloc(384 * 128 * 2);
  _Float16* whT_gru = (_Float16*)alloc(384 * 128 * 2);
  _Float16* wxT_aug = (_Float16*)alloc(384 * 128 * 2);
  _Float16* wsT_aug = (_Float16*)alloc(384 * 128 * 2);
  _Float16* aux1T   = (_Float16*)alloc(112 * 256 * 2);
  _Float16* fc1T    = (_Float16*)alloc(208 * 576 * 2);
  _Float16* fc2T    = (_Float16*)alloc(80 * 224 * 2);

  // ---- weight conversion (f32 KxN -> f16 Npad x Kpad, zero padded)
  convT<<<(384 * 128 + 255) / 256, 256, 0, stream>>>(gru_wx, wxT_gru, 128, 384, 128, 384);
  convT<<<(384 * 128 + 255) / 256, 256, 0, stream>>>(gru_wh, whT_gru, 128, 384, 128, 384);
  convT<<<(384 * 128 + 255) / 256, 256, 0, stream>>>(aug_wx, wxT_aug, 128, 384, 128, 384);
  convT<<<(384 * 128 + 255) / 256, 256, 0, stream>>>(aug_ws, wsT_aug, 128, 384, 128, 384);
  convT<<<(112 * 256 + 255) / 256, 256, 0, stream>>>(aux_w1, aux1T, 256, 100, 256, 112);
  convT<<<(208 * 576 + 255) / 256, 256, 0, stream>>>(fc_w1, fc1T, 576, 200, 576, 208);
  convT<<<(80 * 224 + 255) / 256, 256, 0, stream>>>(fc_w2, fc2T, 200, 80, 224, 80);
  zero_f32<<<1, 32, 0, stream>>>(acc, 1);
  zero_f16<<<(BB * 224 + 255) / 256, 256, 0, stream>>>(h1_h, BB * 224);

  // ---- embeddings
  gather_kernel<<<BB, 128, 0, stream>>>(user, item, cate, item_his, cate_his,
                                        no_m, no_c, ut, it, ct, user_emb,
                                        item_emb, his_emb, his_emb_h, his_sum,
                                        noclk_sum, noclk_his);

  dim3 gblk(32, 4);
  // ---- GRU: xp = his_emb @ wx + b0, then recurrence
  gemm_f16<<<dim3(BB * TT / 16, 6), gblk, 0, stream>>>(
      his_emb_h, wxT_gru, gru_b, gru_xp, nullptr, BB * TT, 128, 384, 384, 0);
  gru_kernel<<<BB / 16, dim3(32, 8), 0, stream>>>(gru_xp, whT_gru, gru_b + 384,
                                                  mask, gru_out, gru_out_h);
  // ---- AUGRU: xp = gru_out @ wx + b, then recurrence (att == 1)
  gemm_f16<<<dim3(BB * TT / 16, 6), gblk, 0, stream>>>(
      gru_out_h, wxT_aug, aug_b, aug_xp, nullptr, BB * TT, 128, 384, 384, 0);
  augru_kernel<<<BB / 16, dim3(32, 8), 0, stream>>>(aug_xp, wsT_aug, mask,
                                                    behavior);
  // ---- auxiliary loss MLP
  aux_build<<<AUXM, 256, 0, stream>>>(gru_out, his_emb, noclk_his, aux_g,
                                      aux_be, clk_in, nclk_in);
  gemm_f16<<<dim3(AUXM / 16, 2), gblk, 0, stream>>>(
      clk_in, aux1T, aux_b1, nullptr, clk_h1, AUXM, 256, 112, 100, 1);
  gemm_f16<<<dim3(AUXM / 16, 2), gblk, 0, stream>>>(
      nclk_in, aux1T, aux_b1, nullptr, nclk_h1, AUXM, 256, 112, 100, 1);
  aux_tail<<<(2 * AUXM + 255) / 256, 256, 0, stream>>>(
      clk_h1, nclk_h1, aux_w2, aux_b2, aux_w3, aux_b3, mask, acc);
  // ---- final FC with DICE (batch-axis stats)
  fc_build<<<(BB * 576 + 255) / 256, 256, 0, stream>>>(
      user_emb, item_emb, his_sum, behavior, noclk_sum, fc_g, fc_be, fcin_h);
  gemm_f16<<<dim3(BB / 16, 4), gblk, 0, stream>>>(
      fcin_h, fc1T, fc_b1, a1, nullptr, BB, 576, 208, 200, 0);
  dice_kernel<<<200, 256, 0, stream>>>(a1, 208, fc_a1, h1_h, 224);
  gemm_f16<<<dim3(BB / 16, 2), gblk, 0, stream>>>(
      h1_h, fc2T, fc_b2, a2, nullptr, BB, 224, 80, 80, 0);
  dice_kernel<<<80, 256, 0, stream>>>(a2, 80, fc_a2, h2_h, 96);
  final_kernel<<<(BB + 255) / 256, 256, 0, stream>>>(h2_h, fc_w3, fc_b3, acc,
                                                     out);
}
